// SigmoidAttention_88837103551418
// MI455X (gfx1250) — matrix operand order repaired and checked by reading, not compiled, not run on hardware
//
#include <hip/hip_runtime.h>
#include <hip/hip_bf16.h>

// ---------------------------------------------------------------------------
// Sigmoid attention (QK-norm, no softmax) for MI455X / gfx1250, bf16 WMMA.
// B=2, L=2048, DIM=1024, H=16, HD=64.
// ---------------------------------------------------------------------------

typedef __bf16 bf16_t;
typedef __attribute__((ext_vector_type(8)))  __bf16 v8bf;
typedef __attribute__((ext_vector_type(16))) __bf16 v16bf;
typedef __attribute__((ext_vector_type(8)))  float  v8f;

union AFrag { v16bf v; v8bf h[2]; };

__device__ __forceinline__ v8f wmma_bf16(v16bf a, v16bf b, v8f c) {
    // D = A(16x32 bf16) * B(32x16 bf16) + C(16x16 f32)
    return __builtin_amdgcn_wmma_f32_16x16x32_bf16(
        /*neg_a=*/false, a, /*neg_b=*/false, b,
        /*c_mod=*/(short)0, c, /*reuse_a=*/false, /*reuse_b=*/false);
}

static constexpr int DIM = 1024;
static constexpr int NH  = 16;
static constexpr int HD  = 64;
static constexpr int LSEQ = 2048;

// ---------------------------------------------------------------------------
// Kernel 1: Y = X @ W^T  (W stored [out,in]), N-tile of 64 == one head.
// Optional per-row LayerNorm over the 64-wide head dim, then *scale,
// output bf16 head-major [B][H][L][64].
// Block: 256 threads = 8 waves; M-tile 128, N-tile 64; K chunks of 32.
// ---------------------------------------------------------------------------
__global__ __launch_bounds__(256)
void proj_ln_kernel(const float* __restrict__ x, const float* __restrict__ W,
                    const float* __restrict__ gamma, const float* __restrict__ beta,
                    bf16_t* __restrict__ outHM, int applyLN, float scale)
{
    __shared__ bf16_t Xs[128 * 32];   // 8 KB
    __shared__ bf16_t Ws[64 * 32];    // 4 KB

    const int tid  = threadIdx.x;
    const int wave = tid >> 5;
    const int lane = tid & 31;
    const int l16  = lane & 15;
    const int hi   = lane >> 4;          // 0: rows 0-7 / K lo ; 1: rows 8-15 / K hi
    const int m0   = blockIdx.x * 128;   // row tile over B*L
    const int n0   = blockIdx.y * 64;    // head * 64

    v8f zero = {0.f,0.f,0.f,0.f,0.f,0.f,0.f,0.f};
    v8f acc[4] = {zero, zero, zero, zero};

    for (int k0 = 0; k0 < DIM; k0 += 32) {
        // ---- cooperative stage: X tile 128x32 fp32 -> bf16 LDS (coalesced)
        #pragma unroll
        for (int i = 0; i < 16; ++i) {
            int idx = tid + i * 256;
            int r = idx >> 5, kk = idx & 31;
            Xs[idx] = (bf16_t)x[(size_t)(m0 + r) * DIM + k0 + kk];
        }
        // ---- stage W tile 64x32
        #pragma unroll
        for (int i = 0; i < 8; ++i) {
            int idx = tid + i * 256;
            int r = idx >> 5, kk = idx & 31;
            Ws[idx] = (bf16_t)W[(size_t)(n0 + r) * DIM + k0 + kk];
        }
        __syncthreads();

        // ---- A fragment: row = wave*16 + l16, K runs {0..15}/{16..31}
        AFrag a;
        const bf16_t* ap = &Xs[(wave * 16 + l16) * 32 + (hi ? 16 : 0)];
        a.h[0] = *(const v8bf*)ap;
        a.h[1] = *(const v8bf*)(ap + 8);

        #pragma unroll
        for (int t = 0; t < 4; ++t) {
            // B fragment: column n = t*16+l16 (a row of W), 16 contiguous K
            AFrag b;
            b.v = *(const v16bf*)&Ws[(t * 16 + l16) * 32 + (hi ? 16 : 0)];
            acc[t] = wmma_bf16(a.v, b.v, acc[t]);
        }
        __syncthreads();
    }

    // ---- optional LayerNorm over the 64 columns of each row -----------------
    float mean[8], rstd[8];
    if (applyLN) {
        #pragma unroll
        for (int r = 0; r < 8; ++r) {
            float s = 0.f, sq = 0.f;
            #pragma unroll
            for (int t = 0; t < 4; ++t) { float v = acc[t][r]; s += v; sq += v * v; }
            // reduce across the 16 lanes of this half-wave (masks < 16 stay in-group)
            #pragma unroll
            for (int m = 1; m < 16; m <<= 1) {
                s  += __shfl_xor(s,  m, 32);
                sq += __shfl_xor(sq, m, 32);
            }
            float mu  = s * (1.f / 64.f);
            float var = sq * (1.f / 64.f) - mu * mu;
            mean[r] = mu;
            rstd[r] = rsqrtf(var + 1e-5f);
        }
    }
    float g[4], bt[4];
    #pragma unroll
    for (int t = 0; t < 4; ++t) {
        g[t]  = applyLN ? gamma[t * 16 + l16] : 1.f;
        bt[t] = applyLN ? beta[t * 16 + l16]  : 0.f;
    }

    // ---- epilogue: bf16 head-major [B][H][L][64]
    const int h = blockIdx.y;
    #pragma unroll
    for (int t = 0; t < 4; ++t) {
        #pragma unroll
        for (int r = 0; r < 8; ++r) {
            float v = acc[t][r];
            if (applyLN) v = (v - mean[r]) * rstd[r] * g[t] + bt[t];
            v *= scale;
            int m  = m0 + wave * 16 + hi * 8 + r;   // global row over B*L
            int bb = m >> 11;                       // / 2048
            int ll = m & 2047;
            size_t off = ((((size_t)bb * NH + h) * LSEQ) + ll) * HD + t * 16 + l16;
            outHM[off] = (bf16_t)v;
        }
    }
}

// ---------------------------------------------------------------------------
// Kernel 2: O[b,h,q,:] = sum_k sigmoid(Q.K^T) * V   (no normalization)
// Block: 256 thr = 8 waves; per block: one (b,h), 128 queries.
// Streams keys in tiles of 32. Writes fp32 X at [b][q][h*64+d].
// ---------------------------------------------------------------------------
__global__ __launch_bounds__(256)
void sigmoid_attn_kernel(const bf16_t* __restrict__ Qh, const bf16_t* __restrict__ Kh,
                         const bf16_t* __restrict__ Vh, float* __restrict__ X)
{
    __shared__ bf16_t Vt[HD * 32];        // V^T tile [d][key], 4 KB
    __shared__ bf16_t Ps[8][16 * 32];     // per-wave sigmoid probs, 8 KB

    const int tid  = threadIdx.x;
    const int wave = tid >> 5;
    const int lane = tid & 31;
    const int l16  = lane & 15;
    const int hi   = lane >> 4;
    const int q0   = blockIdx.x * 128;
    const int h    = blockIdx.y;
    const int b    = blockIdx.z;
    const size_t headBase = ((size_t)b * NH + h) * (size_t)LSEQ * HD;

    // persistent Q A-fragments (2 chunks of K=32 over head_dim=64), direct global
    AFrag qa[2];
    {
        const bf16_t* qrow = Qh + headBase + (size_t)(q0 + wave * 16 + l16) * HD;
        #pragma unroll
        for (int c = 0; c < 2; ++c) {
            const bf16_t* p = qrow + 32 * c + (hi ? 16 : 0);
            qa[c].h[0] = *(const v8bf*)p;
            qa[c].h[1] = *(const v8bf*)(p + 8);
        }
    }

    v8f zero = {0.f,0.f,0.f,0.f,0.f,0.f,0.f,0.f};
    v8f o[4] = {zero, zero, zero, zero};

    for (int k0 = 0; k0 < LSEQ; k0 += 32) {
        // ---- stage V tile transposed into LDS: Vt[d][key]
        {
            const bf16_t* vsrc = Vh + headBase + (size_t)k0 * HD + (size_t)tid * 8;
            v8bf vv = *(const v8bf*)vsrc;          // coalesced 16B read
            int key = tid >> 3;                    // (tid*8)/64
            int d0  = (tid * 8) & 63;
            #pragma unroll
            for (int j = 0; j < 8; ++j) Vt[(d0 + j) * 32 + key] = vv[j];
        }
        if (k0 + 32 < LSEQ) {   // hint next tiles into cache (global_prefetch_b8)
            __builtin_prefetch(Kh + headBase + (size_t)(k0 + 32) * HD + tid * 8, 0, 1);
            __builtin_prefetch(Vh + headBase + (size_t)(k0 + 32) * HD + tid * 8, 0, 1);
        }
        __syncthreads();

        // ---- S (16q x 32keys) = Q . K^T ; K B-frags straight from global
        v8f s[2] = {zero, zero};
        #pragma unroll
        for (int c = 0; c < 2; ++c) {
            #pragma unroll
            for (int t = 0; t < 2; ++t) {
                const bf16_t* kp = Kh + headBase
                                 + (size_t)(k0 + t * 16 + l16) * HD
                                 + 32 * c + (hi ? 16 : 0);
                AFrag kb; kb.v = *(const v16bf*)kp;
                s[t] = wmma_bf16(qa[c].v, kb.v, s[t]);
            }
        }

        // ---- sigmoid, restage through LDS to get A-layout for P
        #pragma unroll
        for (int t = 0; t < 2; ++t) {
            #pragma unroll
            for (int r = 0; r < 8; ++r) {
                float p = 1.f / (1.f + __expf(-s[t][r]));
                Ps[wave][(hi * 8 + r) * 32 + t * 16 + l16] = (bf16_t)p;
            }
        }
        AFrag pa;
        const bf16_t* pp = &Ps[wave][l16 * 32 + (hi ? 16 : 0)];
        pa.h[0] = *(const v8bf*)pp;
        pa.h[1] = *(const v8bf*)(pp + 8);

        // ---- O += P (16x32) . V (32keys x 64d)
        #pragma unroll
        for (int t = 0; t < 4; ++t) {
            AFrag vb;
            vb.v = *(const v16bf*)&Vt[(t * 16 + l16) * 32 + (hi ? 16 : 0)];
            o[t] = wmma_bf16(pa.v, vb.v, o[t]);
        }
        __syncthreads();
    }

    // ---- store fp32 X[b][q][h*64+d]  (== transpose(0,2,1,3).reshape)
    #pragma unroll
    for (int t = 0; t < 4; ++t) {
        #pragma unroll
        for (int r = 0; r < 8; ++r) {
            int q = q0 + wave * 16 + hi * 8 + r;
            X[((size_t)b * LSEQ + q) * DIM + h * HD + t * 16 + l16] = o[t][r];
        }
    }
}

// ---------------------------------------------------------------------------
// Kernel 3: out = X @ Wo^T + bo   (fp32 in ws, fp32 out)
// ---------------------------------------------------------------------------
__global__ __launch_bounds__(256)
void out_proj_kernel(const float* __restrict__ X, const float* __restrict__ Wo,
                     const float* __restrict__ bo, float* __restrict__ out)
{
    __shared__ bf16_t Xs[128 * 32];
    __shared__ bf16_t Ws[64 * 32];

    const int tid  = threadIdx.x;
    const int wave = tid >> 5;
    const int lane = tid & 31;
    const int l16  = lane & 15;
    const int hi   = lane >> 4;
    const int m0   = blockIdx.x * 128;
    const int n0   = blockIdx.y * 64;

    v8f zero = {0.f,0.f,0.f,0.f,0.f,0.f,0.f,0.f};
    v8f acc[4] = {zero, zero, zero, zero};

    for (int k0 = 0; k0 < DIM; k0 += 32) {
        #pragma unroll
        for (int i = 0; i < 16; ++i) {
            int idx = tid + i * 256;
            int r = idx >> 5, kk = idx & 31;
            Xs[idx] = (bf16_t)X[(size_t)(m0 + r) * DIM + k0 + kk];
        }
        #pragma unroll
        for (int i = 0; i < 8; ++i) {
            int idx = tid + i * 256;
            int r = idx >> 5, kk = idx & 31;
            Ws[idx] = (bf16_t)Wo[(size_t)(n0 + r) * DIM + k0 + kk];
        }
        __syncthreads();

        AFrag a;
        const bf16_t* ap = &Xs[(wave * 16 + l16) * 32 + (hi ? 16 : 0)];
        a.h[0] = *(const v8bf*)ap;
        a.h[1] = *(const v8bf*)(ap + 8);
        #pragma unroll
        for (int t = 0; t < 4; ++t) {
            AFrag bfr;
            bfr.v = *(const v16bf*)&Ws[(t * 16 + l16) * 32 + (hi ? 16 : 0)];
            acc[t] = wmma_bf16(a.v, bfr.v, acc[t]);
        }
        __syncthreads();
    }

    #pragma unroll
    for (int t = 0; t < 4; ++t) {
        float bias = bo[n0 + t * 16 + l16];
        #pragma unroll
        for (int r = 0; r < 8; ++r) {
            int m = m0 + wave * 16 + hi * 8 + r;
            out[(size_t)m * DIM + n0 + t * 16 + l16] = acc[t][r] + bias;
        }
    }
}

// ---------------------------------------------------------------------------
// Host launcher
// ---------------------------------------------------------------------------
extern "C" void kernel_launch(void* const* d_in, const int* in_sizes, int n_in,
                              void* d_out, int out_size, void* d_ws, size_t ws_size,
                              hipStream_t stream)
{
    (void)in_sizes; (void)n_in; (void)out_size; (void)ws_size;
    const float* q  = (const float*)d_in[0];
    const float* k  = (const float*)d_in[1];
    const float* v  = (const float*)d_in[2];
    const float* Wq = (const float*)d_in[3];
    const float* Wk = (const float*)d_in[4];
    const float* Wv = (const float*)d_in[5];
    const float* gq = (const float*)d_in[6];
    const float* bq = (const float*)d_in[7];
    const float* gk = (const float*)d_in[8];
    const float* bk = (const float*)d_in[9];
    const float* Wo = (const float*)d_in[10];
    const float* bo = (const float*)d_in[11];
    float* out = (float*)d_out;

    // workspace layout: Qh(8MB bf16) | Kh(8MB) | Vh(8MB) | X(16MB fp32)
    char* ws = (char*)d_ws;
    bf16_t* Qh = (bf16_t*)(ws);
    bf16_t* Kh = (bf16_t*)(ws + (size_t)(8u << 20));
    bf16_t* Vh = (bf16_t*)(ws + (size_t)(16u << 20));
    float*  Xa = (float*) (ws + (size_t)(24u << 20));

    dim3 blk(256);
    dim3 gProj(32, 16);          // (B*L)/128 x heads
    const float SCALE = 0.125f;  // 64^-0.5

    proj_ln_kernel<<<gProj, blk, 0, stream>>>(q, Wq, gq, bq, Qh, 1, SCALE);
    proj_ln_kernel<<<gProj, blk, 0, stream>>>(k, Wk, gk, bk, Kh, 1, 1.0f);
    proj_ln_kernel<<<gProj, blk, 0, stream>>>(v, Wv, gq, bq, Vh, 0, 1.0f); // LN off

    dim3 gAttn(16, 16, 2);       // L/128 x heads x batch
    sigmoid_attn_kernel<<<gAttn, blk, 0, stream>>>(Qh, Kh, Vh, Xa);

    out_proj_kernel<<<gProj, blk, 0, stream>>>(Xa, Wo, bo, out);
}
